// OrthogonalBasisFFN_87677462380610
// MI455X (gfx1250) — matrix operand order repaired and checked by reading, not compiled, not run on hardware
//
#include <hip/hip_runtime.h>
#include <math.h>

typedef __bf16 bf16_t;
typedef bf16_t v16bf __attribute__((ext_vector_type(16)));
typedef float  v8f   __attribute__((ext_vector_type(8)));

#define T_TOTAL   16384   // B*S
#define D_IN      1024
#define DFF_      4096
#define NB_       8
#define R_        64
#define NBR_      512     // NB*R
#define KSEL      8

#define TM        16      // tokens per workgroup
#define XS_STRIDE 1032    // 1024 + 8 bf16 pad -> row offset = 4 dwords mod 64
#define HN_STRIDE 520     // 512 + 8
#define HA_STRIDE 68      // 64 + 4 (f32)
#define H2_STRIDE 264     // 256 + 8
#define FC        256     // DFF chunk per iteration

// ---------------- bf16 split helpers (RNE) ----------------
__device__ __forceinline__ unsigned short f2bf(float f) {
    unsigned u = __float_as_uint(f);
    u += 0x7fffu + ((u >> 16) & 1u);
    return (unsigned short)(u >> 16);
}
__device__ __forceinline__ float bf2f(unsigned short h) {
    return __uint_as_float((unsigned)h << 16);
}

struct B32x2 { uint4 a, b; };
// Load 16 bf16 for a WMMA operand: 16B at p, 16B at p+off2 (off2 in elements).
// A-operand and B-operand: off2=8 (contiguous 32B).
__device__ __forceinline__ v16bf ld_bf16x16(const unsigned short* p, int off2) {
    B32x2 t;
    t.a = *(const uint4*)(p);
    t.b = *(const uint4*)(p + off2);
    return __builtin_bit_cast(v16bf, t);
}

__device__ __forceinline__ v8f wmma_bf16(v16bf a, v16bf b, v8f c) {
    return __builtin_amdgcn_wmma_f32_16x16x32_bf16(
        /*neg_a=*/false, a, /*neg_b=*/false, b,
        /*c_mod=*/(short)0, c, /*reuse_a=*/false, /*reuse_b=*/false);
}

// ---------------------------------------------------------------------------
// Prep 1: transpose [rows][cols] f32 -> hi/lo bf16 [cols][rows] (batched on z)
// ---------------------------------------------------------------------------
__global__ __launch_bounds__(256)
void transpose_split_kernel(const float* __restrict__ in,
                            unsigned short* __restrict__ outH,
                            unsigned short* __restrict__ outL,
                            int rows, int cols) {
    __shared__ float tile[32][33];
    const size_t bofs = (size_t)blockIdx.z * rows * cols;
    in += bofs; outH += bofs; outL += bofs;
    const int c0 = blockIdx.x * 32, r0 = blockIdx.y * 32;
    const int tx = threadIdx.x, ty = threadIdx.y;
    #pragma unroll
    for (int i = 0; i < 32; i += 8)
        tile[ty + i][tx] = in[(size_t)(r0 + ty + i) * cols + c0 + tx];
    __syncthreads();
    #pragma unroll
    for (int i = 0; i < 32; i += 8) {
        const float v = tile[tx][ty + i];
        const unsigned short h = f2bf(v);
        const unsigned short l = f2bf(v - bf2f(h));
        const size_t o = (size_t)(c0 + ty + i) * rows + r0 + tx;
        outH[o] = h; outL[o] = l;
    }
}

// ---------------------------------------------------------------------------
// Prep 2: elementwise f32 -> hi/lo bf16 (layout preserved), float4 granular
// ---------------------------------------------------------------------------
__global__ __launch_bounds__(256)
void split_conv_kernel(const float* __restrict__ in,
                       unsigned short* __restrict__ outH,
                       unsigned short* __restrict__ outL, int n4) {
    const int i = blockIdx.x * 256 + threadIdx.x;
    if (i >= n4) return;
    const float4 v = ((const float4*)in)[i];
    ushort4 h, l;
    h.x = f2bf(v.x); l.x = f2bf(v.x - bf2f(h.x));
    h.y = f2bf(v.y); l.y = f2bf(v.y - bf2f(h.y));
    h.z = f2bf(v.z); l.z = f2bf(v.z - bf2f(h.z));
    h.w = f2bf(v.w); l.w = f2bf(v.w - bf2f(h.w));
    ((ushort4*)outH)[i] = h;
    ((ushort4*)outL)[i] = l;
}

// ---------------------------------------------------------------------------
// Main: fused OrthogonalBasisFFN, bf16 split-3 WMMA. 8 waves / 16 tokens.
// ---------------------------------------------------------------------------
__global__ __launch_bounds__(256)
void obffn_main_kernel(const float* __restrict__ x,
                       const int*   __restrict__ neuron_idx,
                       const float* __restrict__ neuron_weights,
                       const float* __restrict__ neuron_coords,   // [N][NB]
                       const unsigned short* __restrict__ AhT,    // [NB*R][D] bf16 hi
                       const unsigned short* __restrict__ AlT,    // [NB*R][D] bf16 lo
                       const unsigned short* __restrict__ BhT,    // [DFF][NBR] bf16 hi
                       const unsigned short* __restrict__ BlT,    // [DFF][NBR] bf16 lo
                       const unsigned short* __restrict__ WhT,    // [D][DFF] bf16 hi
                       const unsigned short* __restrict__ WlT,    // [D][DFF] bf16 lo
                       const float* __restrict__ W_down_b,        // [D]
                       float*       __restrict__ out)             // [T][D]
{
    __shared__ __align__(16) unsigned short XhS[TM * XS_STRIDE];  // 33.0 KB
    __shared__ __align__(16) unsigned short XlS[TM * XS_STRIDE];  // 33.0 KB
    __shared__ float cS[TM * NB_];                                //  0.5 KB
    __shared__ float hApart[NB_ * TM * HA_STRIDE];                // 34.8 KB
    __shared__ __align__(16) unsigned short hnHS[TM * HN_STRIDE]; // 16.6 KB
    __shared__ __align__(16) unsigned short hnLS[TM * HN_STRIDE]; // 16.6 KB
    __shared__ __align__(16) unsigned short h2HS[TM * H2_STRIDE]; //  8.4 KB
    __shared__ __align__(16) unsigned short h2LS[TM * H2_STRIDE]; //  8.4 KB

    const int tid   = threadIdx.x;
    const int wave  = tid >> 5;          // 0..7
    const int lane  = tid & 31;
    const int lhalf = lane & 15;         // M row / N col within tile
    const int aoff  = (lane >> 4) << 4;  // A-operand K sub-offset: 0 or 16
    const int boff  = (lane >> 4) << 4;  // B-operand K sub-offset: 0 or 16
    const int mbase = (lane >> 4) << 3;  // C/D rows 0..7 or 8..15
    const int t0    = blockIdx.x * TM;

    // ---- Phase A0: token coords c[t][n]
    if (tid < TM * NB_) {
        const int tt = tid >> 3;
        const int n  = tid & 7;
        float acc = 0.f;
        #pragma unroll
        for (int k = 0; k < KSEL; ++k) {
            const int gidx = neuron_idx[(size_t)(t0 + tt) * KSEL + k];
            acc += neuron_weights[(size_t)(t0 + tt) * KSEL + k] *
                   neuron_coords[(size_t)gidx * NB_ + n];
        }
        cS[tt * NB_ + n] = acc;
    }

    // ---- Phase A1: stage x tile [16][1024] as hi/lo bf16 in LDS
    for (int i = tid; i < TM * D_IN / 4; i += 256) {
        const int row = i >> 8;
        const int c4  = i & 255;
        const float4 v = ((const float4*)(x + (size_t)(t0 + row) * D_IN))[c4];
        ushort4 h, l;
        h.x = f2bf(v.x); l.x = f2bf(v.x - bf2f(h.x));
        h.y = f2bf(v.y); l.y = f2bf(v.y - bf2f(h.y));
        h.z = f2bf(v.z); l.z = f2bf(v.z - bf2f(h.z));
        h.w = f2bf(v.w); l.w = f2bf(v.w - bf2f(h.w));
        *(ushort4*)&XhS[row * XS_STRIDE + c4 * 4] = h;
        *(ushort4*)&XlS[row * XS_STRIDE + c4 * 4] = l;
    }
    __syncthreads();

    // ---- Phase B: GEMM1 (wave w -> basis n=w): hA[16 tok][64 r], K=1024
    {
        v8f acc[4];
        #pragma unroll
        for (int i = 0; i < 4; ++i) acc[i] = 0.0f;
        const unsigned short* xhp = &XhS[lhalf * XS_STRIDE + aoff];
        const unsigned short* xlp = &XlS[lhalf * XS_STRIDE + aoff];
        const size_t abase = (size_t)(wave * 64 + lhalf) * D_IN + boff;
        for (int k0 = 0; k0 < D_IN; k0 += 32) {
            const v16bf xh = ld_bf16x16(xhp + k0, 8);
            const v16bf xl = ld_bf16x16(xlp + k0, 8);
            #pragma unroll
            for (int rt = 0; rt < 4; ++rt) {
                const size_t o = abase + (size_t)rt * 16 * D_IN + k0;
                const v16bf bh = ld_bf16x16(AhT + o, 8);
                const v16bf bl = ld_bf16x16(AlT + o, 8);
                acc[rt] = wmma_bf16(xh, bh, acc[rt]);
                acc[rt] = wmma_bf16(xh, bl, acc[rt]);
                acc[rt] = wmma_bf16(xl, bh, acc[rt]);
            }
        }
        // scale by c[t][wave], store f32 partials
        #pragma unroll
        for (int rt = 0; rt < 4; ++rt) {
            #pragma unroll
            for (int v = 0; v < 8; ++v) {
                const int t = mbase + v;
                hApart[(wave * TM + t) * HA_STRIDE + rt * 16 + lhalf] =
                    acc[rt][v] * cS[t * NB_ + wave];
            }
        }
    }
    __syncthreads();

    // ---- Phase B2: h = sum_n partials; hn = outer(c, h) as hi/lo bf16
    for (int i = tid; i < TM * R_; i += 256) {
        const int t = i >> 6;
        const int r = i & 63;
        float h = 0.f;
        #pragma unroll
        for (int n = 0; n < NB_; ++n)
            h += hApart[(n * TM + t) * HA_STRIDE + r];
        #pragma unroll
        for (int n = 0; n < NB_; ++n) {
            const float val = cS[t * NB_ + n] * h;
            const unsigned short vh = f2bf(val);
            const unsigned short vl = f2bf(val - bf2f(vh));
            const int o = t * HN_STRIDE + n * R_ + r;
            hnHS[o] = vh; hnLS[o] = vl;
        }
    }
    __syncthreads();

    // ---- Phase C: chunks of DFF: h2 = gelu(hn@B); out += h2 @ Wdown^T
    v8f oacc[8];
    #pragma unroll
    for (int i = 0; i < 8; ++i) oacc[i] = 0.0f;

    for (int fcb = 0; fcb < DFF_ / FC; ++fcb) {
        const int F0 = fcb * FC;

        // GEMM2: wave w -> f-cols [F0+w*32, +32) as 2 tiles, K = 512
        v8f h0 = {}, h1 = {};
        {
            const unsigned short* hh = &hnHS[lhalf * HN_STRIDE + aoff];
            const unsigned short* hl = &hnLS[lhalf * HN_STRIDE + aoff];
            const size_t f0 = (size_t)(F0 + wave * 32 + lhalf) * NBR_ + boff;
            const size_t f1 = f0 + (size_t)16 * NBR_;
            for (int k0 = 0; k0 < NBR_; k0 += 32) {
                const v16bf ah  = ld_bf16x16(hh + k0, 8);
                const v16bf al  = ld_bf16x16(hl + k0, 8);
                const v16bf b0h = ld_bf16x16(BhT + f0 + k0, 8);
                const v16bf b0l = ld_bf16x16(BlT + f0 + k0, 8);
                const v16bf b1h = ld_bf16x16(BhT + f1 + k0, 8);
                const v16bf b1l = ld_bf16x16(BlT + f1 + k0, 8);
                h0 = wmma_bf16(ah, b0h, h0);
                h0 = wmma_bf16(ah, b0l, h0);
                h0 = wmma_bf16(al, b0h, h0);
                h1 = wmma_bf16(ah, b1h, h1);
                h1 = wmma_bf16(ah, b1l, h1);
                h1 = wmma_bf16(al, b1h, h1);
            }
        }
        // exact-erf gelu, split-store chunk to LDS
        #pragma unroll
        for (int v = 0; v < 8; ++v) {
            const int rowo = (mbase + v) * H2_STRIDE + wave * 32 + lhalf;
            const float z0 = h0[v];
            const float g0 = 0.5f * z0 * (1.0f + erff(z0 * 0.70710678f));
            const unsigned short g0h = f2bf(g0);
            h2HS[rowo] = g0h; h2LS[rowo] = f2bf(g0 - bf2f(g0h));
            const float z1 = h1[v];
            const float g1 = 0.5f * z1 * (1.0f + erff(z1 * 0.70710678f));
            const unsigned short g1h = f2bf(g1);
            h2HS[rowo + 16] = g1h; h2LS[rowo + 16] = f2bf(g1 - bf2f(g1h));
        }
        __syncthreads();

        // GEMM3: wave w owns d-cols [w*128, +128): 8 tiles, K = FC
        {
            const unsigned short* p2h = &h2HS[lhalf * H2_STRIDE + aoff];
            const unsigned short* p2l = &h2LS[lhalf * H2_STRIDE + aoff];
            const size_t wbase = (size_t)(wave * 128 + lhalf) * DFF_ + F0 + boff;
            for (int k0 = 0; k0 < FC; k0 += 32) {
                const v16bf ah = ld_bf16x16(p2h + k0, 8);
                const v16bf al = ld_bf16x16(p2l + k0, 8);
                #pragma unroll
                for (int dt = 0; dt < 8; ++dt) {
                    const size_t o = wbase + (size_t)dt * 16 * DFF_ + k0;
                    const v16bf bh = ld_bf16x16(WhT + o, 8);
                    const v16bf bl = ld_bf16x16(WlT + o, 8);
                    oacc[dt] = wmma_bf16(ah, bh, oacc[dt]);
                    oacc[dt] = wmma_bf16(ah, bl, oacc[dt]);
                    oacc[dt] = wmma_bf16(al, bh, oacc[dt]);
                }
            }
        }
        __syncthreads();   // h2 LDS reused next chunk
    }

    // ---- Epilogue: bias + store out tile [16][1024]
    #pragma unroll
    for (int dt = 0; dt < 8; ++dt) {
        const int d = wave * 128 + dt * 16 + lhalf;
        const float bias = W_down_b[d];
        #pragma unroll
        for (int v = 0; v < 8; ++v)
            out[(size_t)(t0 + mbase + v) * D_IN + d] = oacc[dt][v] + bias;
    }
}

// ---------------------------------------------------------------------------
extern "C" void kernel_launch(void* const* d_in, const int* in_sizes, int n_in,
                              void* d_out, int out_size, void* d_ws, size_t ws_size,
                              hipStream_t stream) {
    const float* x              = (const float*)d_in[0];
    const int*   neuron_idx     = (const int*)  d_in[1];
    const float* neuron_weights = (const float*)d_in[2];
    const float* basis_A        = (const float*)d_in[3];
    const float* basis_B        = (const float*)d_in[4];
    const float* neuron_coords  = (const float*)d_in[5];
    const float* W_down_w       = (const float*)d_in[6];
    const float* W_down_b       = (const float*)d_in[7];
    float*       out            = (float*)d_out;

    // ws layout (bytes): AhT 1MB | AlT 1MB | BhT 4MB | BlT 4MB | WhT 8MB | WlT 8MB
    char* ws = (char*)d_ws;
    unsigned short* AhT = (unsigned short*)(ws);
    unsigned short* AlT = (unsigned short*)(ws + (1u << 20));
    unsigned short* BhT = (unsigned short*)(ws + (2u << 20));
    unsigned short* BlT = (unsigned short*)(ws + (6u << 20));
    unsigned short* WhT = (unsigned short*)(ws + (10u << 20));
    unsigned short* WlT = (unsigned short*)(ws + (18u << 20));

    // basis_A [8][1024][64] -> [8][64][1024] hi/lo bf16
    transpose_split_kernel<<<dim3(R_ / 32, D_IN / 32, NB_), dim3(32, 8), 0, stream>>>(
        basis_A, AhT, AlT, D_IN, R_);
    // basis_B [512][4096] -> [4096][512] hi/lo bf16
    transpose_split_kernel<<<dim3(DFF_ / 32, NBR_ / 32, 1), dim3(32, 8), 0, stream>>>(
        basis_B, BhT, BlT, NBR_, DFF_);
    // W_down_w [1024][4096] -> hi/lo bf16, layout preserved ( == [d][f] K-major )
    split_conv_kernel<<<dim3(D_IN * DFF_ / 4 / 256), dim3(256), 0, stream>>>(
        W_down_w, WhT, WlT, D_IN * DFF_ / 4);

    obffn_main_kernel<<<dim3(T_TOTAL / TM), dim3(256), 0, stream>>>(
        x, neuron_idx, neuron_weights, neuron_coords,
        AhT, AlT, BhT, BlT, WhT, WlT, W_down_b, out);
}
